// MultiHeadClassifier_53858889891919
// MI455X (gfx1250) — matrix operand order repaired and checked by reading, not compiled, not run on hardware
//
#include <hip/hip_runtime.h>

// MultiHeadClassifier on gfx1250 (MI455X, wave32, WMMA)
// B=32768 rows, F=512 (K), G=16, L=64 -> N=1024 columns, C=1000 classes.
//
// Pipeline (all on `stream`):
//  1) mhc_cvt_w:      W f32 -> f16 hi/lo split in d_ws (2 MB)
//  2) mhc_build_lists: deterministic CSR (class -> column list) in d_ws
//  3) mhc_gemm:       split-f16 WMMA GEMM (3x v_wmma_f32_16x16x32_f16 per K-chunk),
//                     software-pipelined B loads, weighted epilogue staged in LDS,
//                     deterministic gather-by-class, coalesced row writes.

typedef _Float16 h8  __attribute__((ext_vector_type(8)));
typedef _Float16 h16 __attribute__((ext_vector_type(16)));
typedef float    f8  __attribute__((ext_vector_type(8)));

#define B_ROWS 32768
#define F_DIM  512
#define G_NUM  16
#define N_DIM  1024   // G*L
#define C_NUM  1000
#define MT     32     // rows per block
#define A_STR  520    // f16 elems per A row in LDS (padded: conflict-free ds_load_b128)
#define W_STR  1028   // f32 elems per weighted row in LDS (padded: conflict-free b32)

// LDS: union{ A_hi[32][520]+A_lo[32][520] = 66560 B | weighted[32][1028] f32 = 131584 B }
//      + gp[32][16] f32 = 2048 B
#define SMEM_BYTES (131584 + 2048)

static __device__ __forceinline__ h16 cat8(h8 a, h8 b) {
  return __builtin_shufflevector(a, b, 0,1,2,3,4,5,6,7,8,9,10,11,12,13,14,15);
}

// ---------------- prologue 1: W f32 -> f16 hi/lo ----------------
__global__ __launch_bounds__(256) void mhc_cvt_w(const float* __restrict__ W,
                                                 _Float16* __restrict__ whi,
                                                 _Float16* __restrict__ wlo) {
  int i = blockIdx.x * 256 + threadIdx.x;
  if (i < N_DIM * F_DIM) {
    float x = W[i];
    _Float16 hi = (_Float16)x;
    whi[i] = hi;
    wlo[i] = (_Float16)(x - (float)hi);
  }
}

// ---------------- prologue 2: deterministic class->column CSR ----------------
__global__ __launch_bounds__(1024) void mhc_build_lists(const int* __restrict__ label,
                                                        int* __restrict__ offs,
                                                        int* __restrict__ order) {
  __shared__ int lab[1024];
  __shared__ int cnt[1024];
  __shared__ int scn[1024];
  const int t = threadIdx.x;
  lab[t] = label[t];
  cnt[t] = 0;
  __syncthreads();
  atomicAdd(&cnt[lab[t]], 1);
  __syncthreads();
  scn[t] = cnt[t];
  __syncthreads();
  for (int d = 1; d < 1024; d <<= 1) {       // inclusive scan (deterministic)
    int v = (t >= d) ? scn[t - d] : 0;
    __syncthreads();
    scn[t] += v;
    __syncthreads();
  }
  // deterministic rank of column t within its class
  const int myc = lab[t];
  int rank = 0;
  for (int j = 0; j < t; ++j) rank += (lab[j] == myc);
  order[scn[myc] - cnt[myc] + rank] = t;
  if (t < C_NUM) offs[t] = scn[t] - cnt[t];
  if (t == 0) offs[C_NUM] = N_DIM;
}

// ---------------- main GEMM + epilogue ----------------
__global__ __launch_bounds__(256) void mhc_gemm(
    const float* __restrict__ feat, const float* __restrict__ gprob,
    const float* __restrict__ bias,
    const _Float16* __restrict__ whi, const _Float16* __restrict__ wlo,
    const int* __restrict__ offs, const int* __restrict__ order,
    float* __restrict__ out) {
  __shared__ __align__(16) unsigned char smem[SMEM_BYTES];
  _Float16* Ah = (_Float16*)smem;                 // [32][520] f16
  _Float16* Al = (_Float16*)(smem + 33280);       // [32][520] f16
  float*    wst = (float*)smem;                   // aliases A region: [32][1028] f32
  float*    gpl = (float*)(smem + 131584);        // [32][16] f32

  const int tid = threadIdx.x;
  const int m0  = blockIdx.x * MT;

  // group_probs tile -> LDS
  for (int e = tid; e < MT * G_NUM; e += 256)
    gpl[e] = gprob[(size_t)m0 * G_NUM + e];

  // features tile: f32 -> f16 hi/lo into LDS (coalesced)
  for (int e = tid; e < MT * F_DIM; e += 256) {
    const int r = e >> 9, k = e & (F_DIM - 1);
    float x = feat[(size_t)(m0 + r) * F_DIM + k];
    _Float16 hi = (_Float16)x;
    Ah[r * A_STR + k] = hi;
    Al[r * A_STR + k] = (_Float16)(x - (float)hi);
  }
  __syncthreads();

  const int l     = tid & 31;
  const int w     = tid >> 5;       // 8 waves
  const int nlane = l & 15;
  const int lhalf = l >> 4;
  const int m_base  = (w & 1) * 16;       // which 16-row tile
  const int n_wbase = (w >> 1) * 256;     // 256-column slice -> 16 n-tiles

  f8 acc[16];
#pragma unroll
  for (int t = 0; t < 16; ++t)
#pragma unroll
    for (int v = 0; v < 8; ++v) acc[t][v] = 0.0f;

  // B-fragment: lane holds 16 contiguous K of column (n_wbase + t*16 + nlane)
  const size_t boff = (size_t)(n_wbase + nlane) * F_DIM + (size_t)lhalf * 16;

  for (int kc = 0; kc < F_DIM / 32; ++kc) {
    const int k0 = kc * 32;
    // A-fragment from LDS: one contiguous 16-elem run per lane (k = 16*lhalf + s)
    const int ab = (m_base + nlane) * A_STR + k0 + lhalf * 16;
    h16 ahi = cat8(*(const h8*)(Ah + ab), *(const h8*)(Ah + ab + 8));
    h16 alo = cat8(*(const h8*)(Al + ab), *(const h8*)(Al + ab + 8));
    const _Float16* ph = whi + boff + k0;
    const _Float16* pl = wlo + boff + k0;

    // software pipeline: prefetch tile t+1's B fragments before tile t's WMMAs
    h16 bhi = cat8(*(const h8*)(ph), *(const h8*)(ph + 8));
    h16 blo = cat8(*(const h8*)(pl), *(const h8*)(pl + 8));
#pragma unroll
    for (int t = 0; t < 16; ++t) {
      h16 nbhi = bhi, nblo = blo;
      if (t < 15) {
        const size_t to = (size_t)(t + 1) * 16 * F_DIM;
        nbhi = cat8(*(const h8*)(ph + to), *(const h8*)(ph + to + 8));
        nblo = cat8(*(const h8*)(pl + to), *(const h8*)(pl + to + 8));
      }
      // fp32-emulation: hi*hi + lo*hi + hi*lo, f32 accumulate.
      // bhi-only consumers first so blo loads get two WMMAs of latency cover.
      acc[t] = __builtin_amdgcn_wmma_f32_16x16x32_f16(false, ahi, false, bhi,
                                                      (short)0, acc[t], false, false);
      acc[t] = __builtin_amdgcn_wmma_f32_16x16x32_f16(false, alo, false, bhi,
                                                      (short)0, acc[t], false, false);
      acc[t] = __builtin_amdgcn_wmma_f32_16x16x32_f16(false, ahi, false, blo,
                                                      (short)0, acc[t], false, false);
      bhi = nbhi;
      blo = nblo;
    }
  }
  __syncthreads();   // all waves done reading A region

  // stage weighted = gp[m,g] * (logit + bias[n]) into LDS (aliases A region)
#pragma unroll
  for (int t = 0; t < 16; ++t) {
    const int nt = n_wbase + t * 16;
    const int n  = nt + nlane;
    const int g  = nt >> 6;            // uniform per 16-wide tile (64-wide groups)
    const float bv = bias[n];
#pragma unroll
    for (int v = 0; v < 8; ++v) {
      const int m = m_base + v + lhalf * 8;   // C/D layout: VGPR v -> M=v / v+8
      wst[m * W_STR + n] = gpl[m * G_NUM + g] * (acc[t][v] + bv);
    }
  }
  __syncthreads();

  // deterministic gather by class; coalesced output writes (each element once)
  for (int c = tid; c < C_NUM; c += 256) {
    const int o0 = offs[c], o1 = offs[c + 1];
    float s[MT];
#pragma unroll
    for (int m = 0; m < MT; ++m) s[m] = 0.0f;
    for (int j = o0; j < o1; ++j) {
      const int col = order[j];
#pragma unroll
      for (int m = 0; m < MT; ++m) s[m] += wst[m * W_STR + col];
    }
#pragma unroll
    for (int m = 0; m < MT; ++m)
      out[(size_t)(m0 + m) * C_NUM + c] = s[m];
  }
}

extern "C" void kernel_launch(void* const* d_in, const int* in_sizes, int n_in,
                              void* d_out, int out_size, void* d_ws, size_t ws_size,
                              hipStream_t stream) {
  const float* feat  = (const float*)d_in[0];
  const float* gprob = (const float*)d_in[1];
  const float* W     = (const float*)d_in[2];
  const float* bias  = (const float*)d_in[3];
  const int*   label = (const int*)d_in[4];
  float* out = (float*)d_out;

  // workspace layout: [W_hi f16 1MB][W_lo f16 1MB][offs 1024 int][order 1024 int]
  _Float16* whi = (_Float16*)d_ws;
  _Float16* wlo = whi + (size_t)N_DIM * F_DIM;
  int* offs  = (int*)((char*)d_ws + (size_t)2 * N_DIM * F_DIM * sizeof(_Float16));
  int* order = offs + 1024;

  mhc_cvt_w<<<(N_DIM * F_DIM + 255) / 256, 256, 0, stream>>>(W, whi, wlo);
  mhc_build_lists<<<1, 1024, 0, stream>>>(label, offs, order);
  mhc_gemm<<<B_ROWS / MT, 256, 0, stream>>>(feat, gprob, bias, whi, wlo, offs, order, out);
}
